// EGGModel_84679575208355
// MI455X (gfx1250) — matrix operand order repaired and checked by reading, not compiled, not run on hardware
//
#include <hip/hip_runtime.h>
#include <hip/hip_bf16.h>

// ---------------- model dims ----------------
#define BB     32      // batch
#define DD     256     // hidden
#define NLAYER 6
#define MLPD   1024
#define LL     1024    // seq len
#define VV     256     // vocab
#define RING   8       // ring-buffer depth per pipeline level
#define TPB    512     // threads per block (16 waves of 32)
#define NWAVES (TPB/32)

typedef __attribute__((ext_vector_type(16))) __bf16 v16bf;
typedef __attribute__((ext_vector_type(8)))  float  v8f;
typedef __attribute__((ext_vector_type(4)))  unsigned int uint32x4;
typedef __attribute__((ext_vector_type(8)))  int  int32x8;
typedef __attribute__((ext_vector_type(4)))  int  int32x4;

union FragU { v16bf v; uint4 q[2]; };

// A fragment (16x32 bf16, row-major source [rows][K]):
// lane<16: row=lm, K = kt*32 + {0..7} and {16..23};  lane>=16: K = {8..15},{24..31}
__device__ __forceinline__ v16bf load_fragA(const __bf16* base, int row, int K, int kt, int hi) {
    FragU u;
    const __bf16* p = base + (size_t)row * K + kt * 32 + hi * 8;
    u.q[0] = *(const uint4*)(p);
    u.q[1] = *(const uint4*)(p + 16);
    return u.v;
}

// B fragment (32x16 bf16) from W^T stored [N][K] row-major:
// lane<16: col=lm, K = kt*32 + {0..7} and {16..23} ; lane>=16: K = {8..15},{24..31}
__device__ __forceinline__ v16bf load_fragB(const __bf16* baseT, int col, int K, int kt, int hi) {
    FragU u;
    const __bf16* p = baseT + (size_t)col * K + kt * 32 + hi * 8;
    u.q[0] = *(const uint4*)(p);
    u.q[1] = *(const uint4*)(p + 16);
    return u.v;
}

__device__ __forceinline__ v8f zero8() {
    v8f z = {0.f,0.f,0.f,0.f,0.f,0.f,0.f,0.f};
    return z;
}

__device__ __forceinline__ v8f wmma_bf16(v16bf a, v16bf b, v8f c) {
    return __builtin_amdgcn_wmma_f32_16x16x32_bf16(false, a, false, b, (short)0, c, false, false);
}

// ---- Tensor Data Mover: async copy of a contiguous 2-byte-element tile
// (nbytes, multiple of 32) from global memory into LDS. 2D descriptor,
// tile = nelem x 1, groups 2/3 unused (<=2D tensor). Issued by one wave.
__device__ __forceinline__ void tdm_load_to_lds(const void* gsrc, unsigned lds_off,
                                                unsigned nbytes) {
    unsigned nelem = nbytes >> 1;                       // 2-byte elements (<=65535)
    unsigned long long ga = (unsigned long long)(uintptr_t)gsrc;
    uint32x4 g0;
    g0.x = 1u;                                          // count=1, user descriptor
    g0.y = lds_off;                                     // lds_addr (bytes)
    g0.z = (unsigned)(ga & 0xFFFFFFFFu);                // global_addr[31:0]
    g0.w = (unsigned)((ga >> 32) & 0x01FFFFFFu) | (2u << 30);  // addr[56:32], type=2
    int32x8 g1;
    g1[0] = (int)(1u << 16);                            // data_size=1 (2 bytes)
    g1[1] = (int)((nelem & 0xFFFFu) << 16);             // tensor_dim0[15:0]
    g1[2] = (int)(((nelem >> 16) & 0xFFFFu) | (1u << 16)); // dim0[31:16] | tensor_dim1=1
    g1[3] = (int)((nelem & 0xFFFFu) << 16);             // tile_dim0 = nelem
    g1[4] = 1;                                          // tile_dim1=1, tile_dim2=0
    g1[5] = (int)nelem;                                 // tensor_dim0_stride (lo 32)
    g1[6] = 0;
    g1[7] = 0;
    int32x4 z4 = {0, 0, 0, 0};
#if defined(__clang_major__) && (__clang_major__ >= 23)
    int32x8 z8 = {0, 0, 0, 0, 0, 0, 0, 0};
    __builtin_amdgcn_tensor_load_to_lds(g0, g1, z4, z4, z8, 0);
#else
    __builtin_amdgcn_tensor_load_to_lds(g0, g1, z4, z4, 0);
#endif
}

// LayerNorm of 32 rows x 256 cols fp32 -> bf16, rows distributed over waves.
__device__ __forceinline__ void ln_rows(const float* src, __bf16* dst,
                                        const float* g, const float* b,
                                        int wave, int lane) {
    const int c0 = lane * 8;
    for (int row = wave; row < BB; row += NWAVES) {
        const float* sr = src + row * DD;
        float v[8]; float s = 0.f, ss = 0.f;
#pragma unroll
        for (int j = 0; j < 8; ++j) { float f = sr[c0 + j]; v[j] = f; s += f; ss += f * f; }
#pragma unroll
        for (int o = 16; o > 0; o >>= 1) { s += __shfl_xor(s, o); ss += __shfl_xor(ss, o); }
        float mu = s * (1.f / DD);
        float rs = rsqrtf(ss * (1.f / DD) - mu * mu + 1e-5f);
        __bf16* dr = dst + row * DD;
#pragma unroll
        for (int j = 0; j < 8; ++j)
            dr[c0 + j] = (__bf16)((v[j] - mu) * rs * g[c0 + j] + b[c0 + j]);
    }
}

// ---------------- prep kernels ----------------
__global__ __launch_bounds__(256) void embed_kernel(const int* __restrict__ ids,
                                                    const float* __restrict__ etab,
                                                    __bf16* __restrict__ emb) {
    long i = (long)blockIdx.x * blockDim.x + threadIdx.x;
    if (i >= (long)LL * BB * DD) return;
    int d = (int)(i % DD);
    long r = i / DD;
    int b = (int)(r % BB);
    int t = (int)(r / BB);
    int id = ids[b * LL + t];
    emb[i] = (__bf16)etab[(long)id * DD + d];
}

// src: [M][K][N] f32  ->  dst: [M][N][K] bf16  (per-matrix transpose + convert)
__global__ __launch_bounds__(256) void convert_T(const float* __restrict__ src,
                                                 __bf16* __restrict__ dst,
                                                 int K, int N, long total) {
    long i = (long)blockIdx.x * blockDim.x + threadIdx.x;
    if (i >= total) return;
    long kn = (long)K * N;
    long m = i / kn;
    long r = i % kn;
    int k = (int)(r / N);
    int n = (int)(r % N);
    dst[m * kn + (long)n * K + k] = (__bf16)src[i];
}

// ---------------- persistent pipelined kernel ----------------
// blocks 0..5 = layers (wavefront-pipelined over time), block 6 = output head.
#define SMEM_HST 0
#define SMEM_HN  32768
#define SMEM_MB  49152
#define SMEM_UB  65536
#define SMEM_XB  131072
#define SMEM_TOT 147456

__global__ __launch_bounds__(TPB) void mingru_persist(
    const float* __restrict__ states0,
    const float* __restrict__ pbz, const float* __restrict__ pbh,
    const float* __restrict__ pg1, const float* __restrict__ pb1,
    const float* __restrict__ pg2, const float* __restrict__ pb2,
    const float* __restrict__ pbup, const float* __restrict__ pbdown,
    const __bf16* __restrict__ emb, __bf16* __restrict__ ring,
    const __bf16* __restrict__ wzT, const __bf16* __restrict__ whT,
    const __bf16* __restrict__ wupT, const __bf16* __restrict__ wdT,
    const __bf16* __restrict__ headT,
    int* cnt, float* __restrict__ out)
{
    extern __shared__ char smem[];
    float*  hst = (float*)(smem + SMEM_HST);   // [32][256] f32: h_prev -> h_new (persistent)
    __bf16* hn  = (__bf16*)(smem + SMEM_HN);   // [32][256] bf16: LN1 output
    __bf16* mb  = (__bf16*)(smem + SMEM_MB);   // [32][256] bf16: LN2 output
    __bf16* ub  = (__bf16*)(smem + SMEM_UB);   // [32][1024] bf16: MLP hidden
    __bf16* xb  = (__bf16*)(smem + SMEM_XB);   // [32][256] bf16: TDM-staged input tile

    const unsigned xb_lds = (unsigned)(uintptr_t)xb;   // low 32 bits = LDS byte address

    const int tid  = threadIdx.x;
    const int wave = tid >> 5;
    const int lane = tid & 31;
    const int lm   = lane & 15;
    const int hi   = lane >> 4;
    const int bid  = blockIdx.x;

    if (bid < NLAYER) {
        const int l = bid;
        for (int i = tid; i < BB * DD; i += TPB) hst[i] = states0[l * BB * DD + i];
        __syncthreads();

        const __bf16* wz_l = wzT + (size_t)l * DD * (2 * DD);
        const __bf16* wh_l = whT + (size_t)l * DD * (2 * DD);
        const __bf16* wu_l = wupT + (size_t)l * MLPD * DD;
        const __bf16* wd_l = wdT + (size_t)l * DD * MLPD;
        const float* g1l = pg1 + l * DD;  const float* b1l = pb1 + l * DD;
        const float* g2l = pg2 + l * DD;  const float* b2l = pb2 + l * DD;
        const float* bzl = pbz + l * DD;  const float* bhl = pbh + l * DD;
        const float* bul = pbup + l * MLPD;
        const float* bdl = pbdown + l * DD;

        __bf16* ringOut = ring + (size_t)l * RING * BB * DD;               // level l+1
        const __bf16* ringIn = (l == 0) ? (const __bf16*)nullptr
                                        : ring + (size_t)(l - 1) * RING * BB * DD;
        int* prodIn  = &cnt[l * 16];
        int* prodOut = &cnt[(l + 1) * 16];
        int* consIn  = &cnt[(8 + l) * 16];
        int* consOut = &cnt[(8 + l + 1) * 16];

        for (int t = 0; t < LL; ++t) {
            if (tid == 0) {
                if (l > 0)
                    while (__hip_atomic_load(prodIn, __ATOMIC_ACQUIRE, __HIP_MEMORY_SCOPE_AGENT) <= t)
                        __builtin_amdgcn_s_sleep(2);
                if (t >= RING)
                    while (__hip_atomic_load(consOut, __ATOMIC_ACQUIRE, __HIP_MEMORY_SCOPE_AGENT) <= t - RING)
                        __builtin_amdgcn_s_sleep(2);
            }
            __syncthreads();

            const __bf16* xin = (l == 0) ? emb + (size_t)t * BB * DD
                                         : ringIn + (size_t)(t % RING) * BB * DD;

            // async DMA: stage x[t] tile into LDS, overlapped with LN1 below
            if (wave == 0)
                tdm_load_to_lds(xin, xb_lds, BB * DD * 2);

            // ---- LN1 on h_prev (overlaps the TDM transfer) ----
            ln_rows(hst, hn, g1l, b1l, wave, lane);
            if (wave == 0)
                __builtin_amdgcn_s_wait_tensorcnt(0);
            __syncthreads();

            // ---- gate GEMMs: comb=[x|hn] (K=512) -> z, h_tilde; combine into h_new ----
            for (int tile = wave; tile < 32; tile += NWAVES) {
                int mt = tile >> 4, nt = tile & 15;
                int n = nt * 16 + lm;
                int row = mt * 16 + lm;
                v8f az = zero8(), ah = zero8();
#pragma unroll
                for (int kt = 0; kt < 8; ++kt) {               // K 0..255 from x (LDS, TDM-staged)
                    v16bf a  = load_fragA(xb, row, DD, kt, hi);
                    v16bf bz = load_fragB(wz_l, n, 2 * DD, kt, hi);
                    v16bf bh = load_fragB(wh_l, n, 2 * DD, kt, hi);
                    az = wmma_bf16(a, bz, az);
                    ah = wmma_bf16(a, bh, ah);
                }
#pragma unroll
                for (int kt = 0; kt < 8; ++kt) {               // K 256..511 from hn (LDS)
                    v16bf a  = load_fragA(hn, row, DD, kt, hi);
                    v16bf bz = load_fragB(wz_l, n, 2 * DD, kt + 8, hi);
                    v16bf bh = load_fragB(wh_l, n, 2 * DD, kt + 8, hi);
                    az = wmma_bf16(a, bz, az);
                    ah = wmma_bf16(a, bh, ah);
                }
                float zb = bzl[n], hb = bhl[n];
#pragma unroll
                for (int i = 0; i < 8; ++i) {
                    int m = mt * 16 + hi * 8 + i;
                    float z  = 1.f / (1.f + __expf(-(az[i] + zb)));
                    float ht = tanhf(ah[i] + hb);
                    float hv = (float)hn[m * DD + n];
                    hst[m * DD + n] = (1.f - z) * hv + z * ht;  // h_new overwrites h_prev
                }
            }
            __syncthreads();

            if (t == LL - 1) {  // final_states output (after logits region)
                for (int i = tid; i < BB * DD; i += TPB)
                    out[(size_t)BB * LL * VV + (size_t)l * BB * DD + i] = hst[i];
            }

            // ---- LN2 ----
            ln_rows(hst, mb, g2l, b2l, wave, lane);
            __syncthreads();

            // ---- MLP up (+ReLU) ----
            for (int tile = wave; tile < 128; tile += NWAVES) {
                int mt = tile >> 6, nt = tile & 63;
                int n = nt * 16 + lm;
                int row = mt * 16 + lm;
                v8f ac = zero8();
#pragma unroll
                for (int kt = 0; kt < 8; ++kt) {
                    v16bf a = load_fragA(mb, row, DD, kt, hi);
                    v16bf b = load_fragB(wu_l, n, DD, kt, hi);
                    ac = wmma_bf16(a, b, ac);
                }
                float bn = bul[n];
#pragma unroll
                for (int i = 0; i < 8; ++i) {
                    int m = mt * 16 + hi * 8 + i;
                    ub[m * MLPD + n] = (__bf16)fmaxf(ac[i] + bn, 0.f);
                }
            }
            __syncthreads();

            // ---- MLP down + residual -> ring out ----
            __bf16* xout = ringOut + (size_t)(t % RING) * BB * DD;
            for (int tile = wave; tile < 32; tile += NWAVES) {
                int mt = tile >> 4, nt = tile & 15;
                int n = nt * 16 + lm;
                int row = mt * 16 + lm;
                v8f ac = zero8();
#pragma unroll 4
                for (int kt = 0; kt < 32; ++kt) {
                    v16bf a = load_fragA(ub, row, MLPD, kt, hi);
                    v16bf b = load_fragB(wd_l, n, MLPD, kt, hi);
                    ac = wmma_bf16(a, b, ac);
                }
                float bn = bdl[n];
#pragma unroll
                for (int i = 0; i < 8; ++i) {
                    int m = mt * 16 + hi * 8 + i;
                    xout[m * DD + n] = (__bf16)(ac[i] + bn + hst[m * DD + n]);
                }
            }
            __threadfence();
            __syncthreads();
            if (tid == 0)
                __hip_atomic_store(prodOut, t + 1, __ATOMIC_RELEASE, __HIP_MEMORY_SCOPE_AGENT);
            if (l > 0 && tid == TPB - 1)
                __hip_atomic_store(consIn, t + 1, __ATOMIC_RELEASE, __HIP_MEMORY_SCOPE_AGENT);
        }
    } else {
        // ---- head block: logits[b][t][v] = y @ headT ----
        const __bf16* ringIn = ring + (size_t)(NLAYER - 1) * RING * BB * DD;
        int* prodIn = &cnt[NLAYER * 16];
        int* consIn = &cnt[(8 + NLAYER) * 16];
        for (int t = 0; t < LL; ++t) {
            if (tid == 0)
                while (__hip_atomic_load(prodIn, __ATOMIC_ACQUIRE, __HIP_MEMORY_SCOPE_AGENT) <= t)
                    __builtin_amdgcn_s_sleep(2);
            __syncthreads();
            const __bf16* y = ringIn + (size_t)(t % RING) * BB * DD;
            if (wave == 0) {
                tdm_load_to_lds(y, xb_lds, BB * DD * 2);
                __builtin_amdgcn_s_wait_tensorcnt(0);
            }
            __syncthreads();
            for (int tile = wave; tile < 32; tile += NWAVES) {
                int mt = tile >> 4, nt = tile & 15;
                int n = nt * 16 + lm;
                int row = mt * 16 + lm;
                v8f ac = zero8();
#pragma unroll
                for (int kt = 0; kt < 8; ++kt) {
                    v16bf a = load_fragA(xb, row, DD, kt, hi);
                    v16bf b = load_fragB(headT, n, DD, kt, hi);
                    ac = wmma_bf16(a, b, ac);
                }
#pragma unroll
                for (int i = 0; i < 8; ++i) {
                    int m = mt * 16 + hi * 8 + i;       // batch index
                    out[(size_t)m * LL * VV + (size_t)t * VV + n] = ac[i];
                }
            }
            __syncthreads();
            if (tid == 0)
                __hip_atomic_store(consIn, t + 1, __ATOMIC_RELEASE, __HIP_MEMORY_SCOPE_AGENT);
        }
    }
}

// ---------------- host launcher ----------------
extern "C" void kernel_launch(void* const* d_in, const int* in_sizes, int n_in,
                              void* d_out, int out_size, void* d_ws, size_t ws_size,
                              hipStream_t stream) {
    (void)in_sizes; (void)n_in; (void)out_size; (void)ws_size;
    const int*   ids     = (const int*)d_in[0];
    const float* states0 = (const float*)d_in[1];
    const float* etab    = (const float*)d_in[2];
    const float* Wz      = (const float*)d_in[3];
    const float* bz      = (const float*)d_in[4];
    const float* Wh      = (const float*)d_in[5];
    const float* bh      = (const float*)d_in[6];
    const float* g1      = (const float*)d_in[7];
    const float* b1      = (const float*)d_in[8];
    const float* g2      = (const float*)d_in[9];
    const float* b2      = (const float*)d_in[10];
    const float* Wup     = (const float*)d_in[11];
    const float* bup     = (const float*)d_in[12];
    const float* Wdn     = (const float*)d_in[13];
    const float* bdn     = (const float*)d_in[14];
    const float* head    = (const float*)d_in[15];
    float* out = (float*)d_out;

    char* ws = (char*)d_ws;
    size_t off = 0;
    int*    cnt  = (int*)(ws + off);     off += 4096;
    __bf16* emb  = (__bf16*)(ws + off);  off += (size_t)LL * BB * DD * 2;
    __bf16* ring = (__bf16*)(ws + off);  off += (size_t)NLAYER * RING * BB * DD * 2;
    __bf16* wzT  = (__bf16*)(ws + off);  off += (size_t)NLAYER * 2 * DD * DD * 2;
    __bf16* whT  = (__bf16*)(ws + off);  off += (size_t)NLAYER * 2 * DD * DD * 2;
    __bf16* wuT  = (__bf16*)(ws + off);  off += (size_t)NLAYER * DD * MLPD * 2;
    __bf16* wdT  = (__bf16*)(ws + off);  off += (size_t)NLAYER * MLPD * DD * 2;
    __bf16* hT   = (__bf16*)(ws + off);  off += (size_t)DD * VV * 2;

    hipMemsetAsync(cnt, 0, 4096, stream);

    { long n = (long)LL * BB * DD;
      embed_kernel<<<(unsigned)((n + 255) / 256), 256, 0, stream>>>(ids, etab, emb); }
    { long n = (long)NLAYER * 2 * DD * DD;
      convert_T<<<(unsigned)((n + 255) / 256), 256, 0, stream>>>(Wz, wzT, 2 * DD, DD, n); }
    { long n = (long)NLAYER * 2 * DD * DD;
      convert_T<<<(unsigned)((n + 255) / 256), 256, 0, stream>>>(Wh, whT, 2 * DD, DD, n); }
    { long n = (long)NLAYER * DD * MLPD;
      convert_T<<<(unsigned)((n + 255) / 256), 256, 0, stream>>>(Wup, wuT, DD, MLPD, n); }
    { long n = (long)NLAYER * MLPD * DD;
      convert_T<<<(unsigned)((n + 255) / 256), 256, 0, stream>>>(Wdn, wdT, MLPD, DD, n); }
    { long n = (long)DD * VV;
      convert_T<<<(unsigned)((n + 255) / 256), 256, 0, stream>>>(head, hT, DD, VV, n); }

    hipFuncSetAttribute(reinterpret_cast<const void*>(mingru_persist),
                        hipFuncAttributeMaxDynamicSharedMemorySize, SMEM_TOT);
    mingru_persist<<<NLAYER + 1, TPB, SMEM_TOT, stream>>>(
        states0, bz, bh, g1, b1, g2, b2, bup, bdn,
        emb, ring, wzT, whT, wuT, wdT, hT, cnt, out);
}
